// GraphConvBlock_31061203485067
// MI455X (gfx1250) — matrix operand order repaired and checked by reading, not compiled, not run on hardware
//
#include <hip/hip_runtime.h>
#include <hip/hip_bf16.h>
#include <math.h>

#define NN 50000
#define EE 1600000
#define DD 256
#define SS 5
#define LN_EPS 1e-5f

typedef __bf16 bf16_t;
typedef bf16_t v16bf __attribute__((ext_vector_type(16)));
typedef bf16_t v8bf  __attribute__((ext_vector_type(8)));
typedef float  v8f   __attribute__((ext_vector_type(8)));

// ---------------------------------------------------------------------------
// Kernel 1: SpMM  acc = A @ x   (COO edges, atomic scatter-add; acc is L2-hot)
// ---------------------------------------------------------------------------
__global__ void spmm_kernel(const float* __restrict__ x,
                            const int* __restrict__ rows,
                            const int* __restrict__ cols,
                            const float* __restrict__ vals,
                            float* __restrict__ acc) {
    const int lane = threadIdx.x & 31;
    int wid = (blockIdx.x * blockDim.x + threadIdx.x) >> 5;
    const int nw = (gridDim.x * blockDim.x) >> 5;
    for (int e = wid; e < EE; e += nw) {
        // e is wave-uniform: force scalar loads of the edge triple
        int eu = __builtin_amdgcn_readfirstlane(e);
        const int   r = rows[eu];
        const int   c = cols[eu];
        const float v = vals[eu];
        const float4* src = (const float4*)(x + (size_t)c * DD + lane * 8);
        float4 a0 = src[0];
        float4 a1 = src[1];
        float* dst = acc + (size_t)r * DD + lane * 8;
        atomicAdd(dst + 0, v * a0.x);
        atomicAdd(dst + 1, v * a0.y);
        atomicAdd(dst + 2, v * a0.z);
        atomicAdd(dst + 3, v * a0.w);
        atomicAdd(dst + 4, v * a1.x);
        atomicAdd(dst + 5, v * a1.y);
        atomicAdd(dst + 6, v * a1.z);
        atomicAdd(dst + 7, v * a1.w);
    }
}

// ---------------------------------------------------------------------------
// Kernel 2: fused  out = mask * LN(ELU(acc @ W + bias))
// W is staged in LDS pre-swizzled into WMMA B-fragment order:
//   frag (ks,t) at (ks*16+t)*1024 bytes; lane l halves at l*16 and 512+l*16.
// -> per k-step ONE base pointer; all 16 tiles use 16-bit DS immediates only
//    (no VALU near B regs -> no WMMA hazard NOPs), and each ds_load_b128
//    sweeps all 64 banks (16B lane stride).
// k-loop rolled / t-loop unrolled: wmma=16 in the hot loop, bounded pressure.
// amdgpu_waves_per_eu(4): LDS already caps us at 4 waves/SIMD (2 blocks/WGP),
// so raise the VGPR budget from the 128-reg default to 256 -> no spills.
// ---------------------------------------------------------------------------
__attribute__((amdgpu_waves_per_eu(4)))
__global__ void __launch_bounds__(256)
gemm_ln_kernel(const float* __restrict__ acc,
               const float* __restrict__ W,
               const float* __restrict__ bias,
               const float* __restrict__ gamma,
               const float* __restrict__ beta,
               const float* __restrict__ mask,
               float* __restrict__ out) {
    extern __shared__ bf16_t wt[];   // 128 KB: 8 k-steps * 16 tiles * 1 KB frags

    // ---- stage W -> LDS as bf16 B-fragments (coalesced global reads) ----
    for (int idx = threadIdx.x; idx < (DD * DD) / 4; idx += blockDim.x) {
        const int k  = (idx * 4) / DD;     // K index (row of W)
        const int n0 = (idx * 4) % DD;     // N index
        float4 w4 = *(const float4*)(W + (size_t)k * DD + n0);
        const int ks    = k >> 5;          // k-step
        const int kr    = k & 31;
        const int lhalf = (kr >> 3) & 1;   // lanes 16-31 hold K +8
        const int piece = kr >> 4;         // lo/hi 16B half of the lane's frag
        const int e     = kr & 7;          // element within half
        bf16_t vv[4] = {(bf16_t)w4.x, (bf16_t)w4.y, (bf16_t)w4.z, (bf16_t)w4.w};
        #pragma unroll
        for (int i = 0; i < 4; ++i) {
            const int n = n0 + i;
            const int t = n >> 4;
            const int l = (n & 15) + (lhalf << 4);
            // element index (bf16 units): frag*512 + piece*256 + lane*8 + e
            const int ei = ((ks * 16 + t) << 9) + (piece << 8) + (l << 3) + e;
            wt[ei] = vv[i];
        }
    }
    __syncthreads();

    const int lane = threadIdx.x & 31;
    const int half = lane >> 4;      // 0: lanes 0-15, 1: lanes 16-31
    const int col  = lane & 15;
    const int waveid        = threadIdx.x >> 5;
    const int wavesPerBlock = blockDim.x >> 5;
    const int nTiles        = NN / 16;   // 3125 exactly

    const bf16_t* wb = wt + lane * 8;    // lane's slot within every fragment

    auto cvtA = [](float4 f0, float4 f1, float4 f2, float4 f3) -> v16bf {
        v16bf a;
        a[0]  = (bf16_t)f0.x; a[1]  = (bf16_t)f0.y; a[2]  = (bf16_t)f0.z; a[3]  = (bf16_t)f0.w;
        a[4]  = (bf16_t)f1.x; a[5]  = (bf16_t)f1.y; a[6]  = (bf16_t)f1.z; a[7]  = (bf16_t)f1.w;
        a[8]  = (bf16_t)f2.x; a[9]  = (bf16_t)f2.y; a[10] = (bf16_t)f2.z; a[11] = (bf16_t)f2.w;
        a[12] = (bf16_t)f3.x; a[13] = (bf16_t)f3.y; a[14] = (bf16_t)f3.z; a[15] = (bf16_t)f3.w;
        return a;
    };

    for (int mt = blockIdx.x * wavesPerBlock + waveid; mt < nTiles;
         mt += gridDim.x * wavesPerBlock) {
        const int m0 = mt * 16;

        v8f c[16];
        const v8f vzero = {0.f, 0.f, 0.f, 0.f, 0.f, 0.f, 0.f, 0.f};
        #pragma unroll
        for (int t = 0; t < 16; ++t) c[t] = vzero;

        // A rows: lane covers row m0+col, K-chunks {0..7,16..23} (+8 if half)
        const float* arowbase = acc + (size_t)(m0 + col) * DD + half * 8;

        for (int ks = 0; ks < 8; ++ks) {
            const float* ap = arowbase + ks * 32;
            float4 f0 = *(const float4*)(ap);
            float4 f1 = *(const float4*)(ap + 4);
            float4 f2 = *(const float4*)(ap + 16);
            float4 f3 = *(const float4*)(ap + 20);
            v16bf a = cvtA(f0, f1, f2, f3);

            // one base pointer per k-step; tile offsets are pure DS immediates
            const bf16_t* kb = wb + ks * (16 * 512);
            #pragma unroll
            for (int t = 0; t < 16; ++t) {
                v8bf blo = *(const v8bf*)(kb + t * 512);
                v8bf bhi = *(const v8bf*)(kb + t * 512 + 256);
                v16bf b = __builtin_shufflevector(blo, bhi,
                        0, 1, 2, 3, 4, 5, 6, 7, 8, 9, 10, 11, 12, 13, 14, 15);
                c[t] = __builtin_amdgcn_wmma_f32_16x16x32_bf16(
                        false, a, false, b, (short)0, c[t], false, false);
            }
        }

        // ---- epilogue on C fragments: element (row = m0+j+8*half, n = t*16+col)
        float sum[8], sq[8];
        #pragma unroll
        for (int j = 0; j < 8; ++j) { sum[j] = 0.f; sq[j] = 0.f; }

        #pragma unroll
        for (int t = 0; t < 16; ++t) {
            const float bn = bias[t * 16 + col];
            #pragma unroll
            for (int j = 0; j < 8; ++j) {
                float xv = c[t][j] + bn;
                xv = xv > 0.f ? xv : (__expf(xv) - 1.f);   // ELU
                c[t][j] = xv;
                sum[j] += xv;
                sq[j]  += xv * xv;
            }
        }
        // reduce across the 16 lanes of each half (xor of low 4 bits stays in half)
        #pragma unroll
        for (int j = 0; j < 8; ++j) {
            #pragma unroll
            for (int m = 1; m < 16; m <<= 1) {
                sum[j] += __shfl_xor(sum[j], m, 32);
                sq[j]  += __shfl_xor(sq[j],  m, 32);
            }
        }
        float mu[8], rs[8];
        #pragma unroll
        for (int j = 0; j < 8; ++j) {
            mu[j] = sum[j] * (1.0f / DD);
            float var = sq[j] * (1.0f / DD) - mu[j] * mu[j];
            rs[j] = rsqrtf(var + LN_EPS);
        }

        #pragma unroll
        for (int t = 0; t < 16; ++t) {
            const int n = t * 16 + col;
            const float g  = gamma[n];
            const float bt = beta[n];
            float mk[SS];
            #pragma unroll
            for (int s = 0; s < SS; ++s) mk[s] = mask[s * DD + n];
            #pragma unroll
            for (int j = 0; j < 8; ++j) {
                const float y = (c[t][j] - mu[j]) * rs[j] * g + bt;
                const size_t rowoff = (size_t)(m0 + j + 8 * half) * (SS * DD) + n;
                #pragma unroll
                for (int s = 0; s < SS; ++s)
                    out[rowoff + (size_t)s * DD] = y * mk[s];
            }
        }
    }
}

// ---------------------------------------------------------------------------
extern "C" void kernel_launch(void* const* d_in, const int* in_sizes, int n_in,
                              void* d_out, int out_size, void* d_ws, size_t ws_size,
                              hipStream_t stream) {
    const float* x     = (const float*)d_in[0];
    const int*   adj   = (const int*)  d_in[1];   // [2, E]
    const float* vals  = (const float*)d_in[2];
    const float* mask  = (const float*)d_in[3];
    const float* W     = (const float*)d_in[4];
    const float* bias  = (const float*)d_in[5];
    const float* gamma = (const float*)d_in[6];
    const float* beta  = (const float*)d_in[7];
    float* out = (float*)d_out;

    float* accbuf = (float*)d_ws;                 // N*D floats = 51.2 MB (L2-resident)
    const int* rows = adj;
    const int* cols = adj + EE;

    hipMemsetAsync(accbuf, 0, (size_t)NN * DD * sizeof(float), stream);

    spmm_kernel<<<4096, 256, 0, stream>>>(x, rows, cols, vals, accbuf);

    const size_t ldsBytes = 8 * 16 * 1024;        // 128 KB fragment image of W
    gemm_ln_kernel<<<391, 256, ldsBytes, stream>>>(accbuf, W, bias, gamma, beta, mask, out);
}
